// GraphEncoder_20641612824811
// MI455X (gfx1250) — matrix operand order repaired and checked by reading, not compiled, not run on hardware
//
#include <hip/hip_runtime.h>
#include <hip/hip_bf16.h>
#include <math.h>

#define NEG_SLOPE 0.2f
#define EPSF 1e-16f

typedef __bf16 bf16x16 __attribute__((ext_vector_type(16)));
typedef float  f32x8   __attribute__((ext_vector_type(8)));

// ---------------- order-preserving float <-> uint key (for atomic max) ----------
__device__ __forceinline__ unsigned fkey(float f) {
  unsigned u = __float_as_uint(f);
  return (u & 0x80000000u) ? ~u : (u | 0x80000000u);
}
__device__ __forceinline__ float funkey(unsigned k) {
  return (k & 0x80000000u) ? __uint_as_float(k & 0x7fffffffu) : __uint_as_float(~k);
}

// ============================ fragment packing =================================
// Packed layout: frag[( (tile * (K/32) + kk) * 32 + lane ) * 16 + i], 32B per lane.
// A frag (16x32, row-major source): lane L<16 -> row L,  K in {[0..8),[16..24)}
//                                   lane L>=16 -> row L-16, K in {[8..16),[24..32)}
__global__ void pack_A(const float* __restrict__ A, __bf16* __restrict__ Ap,
                       int M, int K) {
  long i = blockIdx.x * (long)blockDim.x + threadIdx.x;
  int  KK = K >> 5;
  long total = (long)(M >> 4) * KK * 32;
  if (i >= total) return;
  int  lane = (int)(i & 31);
  long t = i >> 5;
  int  mt = (int)(t / KK), kk = (int)(t - (long)mt * KK);
  const float* p = A + (size_t)(mt * 16 + (lane & 15)) * K + kk * 32 + ((lane & 16) ? 8 : 0);
  __bf16* op = Ap + i * 16;
#pragma unroll
  for (int j = 0; j < 8; ++j) op[j] = (__bf16)p[j];
#pragma unroll
  for (int j = 0; j < 8; ++j) op[8 + j] = (__bf16)p[16 + j];
}
// B frag (32x16, row-major source K x N): lane L<16 -> col L,  K in {[0..8),[16..24)}
//                                         lane L>=16 -> col L-16, K in {[8..16),[24..32)}
__global__ void pack_B(const float* __restrict__ B, __bf16* __restrict__ Bp,
                       int K, int N) {
  long i = blockIdx.x * (long)blockDim.x + threadIdx.x;
  int  KK = K >> 5;
  long total = (long)(N >> 4) * KK * 32;
  if (i >= total) return;
  int  lane = (int)(i & 31);
  long t = i >> 5;
  int  nt = (int)(t / KK), kk = (int)(t - (long)nt * KK);
  const float* p = B + (size_t)(kk * 32 + ((lane & 16) ? 8 : 0)) * N + nt * 16 + (lane & 15);
  __bf16* op = Bp + i * 16;
#pragma unroll
  for (int j = 0; j < 16; ++j) op[j] = (__bf16)p[(size_t)(j < 8 ? j : j + 8) * N];
}

// ============== WMMA bf16 GEMM on packed fragments, fp32 accumulate ============
template <int K>
__global__ void gemm_wmma_packed(const __bf16* __restrict__ Ap,
                                 const __bf16* __restrict__ Bp,
                                 float* __restrict__ C, int M, int N) {
  constexpr int KK = K / 32;
  int wave = (int)((blockIdx.x * blockDim.x + threadIdx.x) >> 5);
  int lane = threadIdx.x & 31;
  int ntiles = N >> 4;
  int mt = wave / ntiles;
  int nt = wave - mt * ntiles;
  if (mt * 16 >= M) return;  // wave-uniform: EXEC stays all-1s for the WMMAs
  const bf16x16* ap = (const bf16x16*)Ap + (size_t)mt * KK * 32 + lane;
  const bf16x16* bp = (const bf16x16*)Bp + (size_t)nt * KK * 32 + lane;
  f32x8 acc = {0.f, 0.f, 0.f, 0.f, 0.f, 0.f, 0.f, 0.f};
#pragma unroll
  for (int kk = 0; kk < KK; ++kk) {
    acc = __builtin_amdgcn_wmma_f32_16x16x32_bf16(false, ap[(size_t)kk * 32], false,
                                                  bp[(size_t)kk * 32], (short)0, acc,
                                                  false, false);
  }
  // C tile store: VGPR r -> row mt*16 + 8*(lane>>4) + r, col nt*16 + (lane&15)
  float* cp = C + (size_t)(mt * 16 + ((lane >> 4) << 3)) * N + nt * 16 + (lane & 15);
#pragma unroll
  for (int r = 0; r < 8; ++r) cp[(size_t)r * N] = acc[r];
}

// ---------------- utility fills ------------------------------------------------
__global__ void fill_u32(unsigned* __restrict__ p, unsigned v, long n) {
  long i = blockIdx.x * (long)blockDim.x + threadIdx.x;
  if (i < n) p[i] = v;
}

// ---------------- attention dot products: a_src/a_dst [N,H] --------------------
__global__ void attn_dot(const float* __restrict__ h, const float* __restrict__ att_src,
                         const float* __restrict__ att_dst, float* __restrict__ a_src,
                         float* __restrict__ a_dst, int N, int H, int C) {
  long i = blockIdx.x * (long)blockDim.x + threadIdx.x;
  if (i >= (long)N * H) return;
  int n = (int)(i / H), hd = (int)(i - (long)n * H);
  const float* hp = h + ((size_t)n * H + hd) * C;
  const float* as = att_src + (size_t)hd * C;
  const float* ad = att_dst + (size_t)hd * C;
  float s = 0.f, d = 0.f;
  for (int c = 0; c < C; ++c) { float v = hp[c]; s += v * as[c]; d += v * ad[c]; }
  a_src[i] = s; a_dst[i] = d;
}

// ---------------- edge softmax passes (self loops appended logically) ----------
__device__ __forceinline__ void edge_sd(const long long* __restrict__ ei, long E, long e,
                                        int& s, int& d) {
  if (e < E) { s = (int)ei[e]; d = (int)ei[E + e]; }
  else       { int n = (int)(e - E); s = n; d = n; }
}
__device__ __forceinline__ float edge_raw(const float* a_src, const float* a_dst,
                                          int s, int d, int H, int hd) {
  float r = a_src[(size_t)s * H + hd] + a_dst[(size_t)d * H + hd];
  return r > 0.f ? r : NEG_SLOPE * r;
}

__global__ void edge_attn_max(const long long* __restrict__ ei, long E, int N, int H,
                              const float* __restrict__ a_src,
                              const float* __restrict__ a_dst,
                              unsigned* __restrict__ nmaxk) {
  long i = blockIdx.x * (long)blockDim.x + threadIdx.x;
  long total = (E + (long)N) * H;
  if (i >= total) return;
  long e = i / H; int hd = (int)(i - e * H);
  int s, d; edge_sd(ei, E, e, s, d);
  atomicMax(&nmaxk[(size_t)d * H + hd], fkey(edge_raw(a_src, a_dst, s, d, H, hd)));
}

__global__ void edge_attn_sum(const long long* __restrict__ ei, long E, int N, int H,
                              const float* __restrict__ a_src,
                              const float* __restrict__ a_dst,
                              const unsigned* __restrict__ nmaxk,
                              float* __restrict__ nsum) {
  long i = blockIdx.x * (long)blockDim.x + threadIdx.x;
  long total = (E + (long)N) * H;
  if (i >= total) return;
  long e = i / H; int hd = (int)(i - e * H);
  int s, d; edge_sd(ei, E, e, s, d);
  size_t dh = (size_t)d * H + hd;
  atomicAdd(&nsum[dh], expf(edge_raw(a_src, a_dst, s, d, H, hd) - funkey(nmaxk[dh])));
}

__global__ void edge_aggregate(const long long* __restrict__ ei, long E, int N, int H,
                               int C, const float* __restrict__ a_src,
                               const float* __restrict__ a_dst,
                               const unsigned* __restrict__ nmaxk,
                               const float* __restrict__ nsum,
                               const float* __restrict__ h, float* __restrict__ out) {
  long i = blockIdx.x * (long)blockDim.x + threadIdx.x;
  long total = (E + (long)N) * H;
  if (i >= total) return;
  long e = i / H; int hd = (int)(i - e * H);
  int s, d; edge_sd(ei, E, e, s, d);
  size_t dh = (size_t)d * H + hd;
  float w = expf(edge_raw(a_src, a_dst, s, d, H, hd) - funkey(nmaxk[dh])) /
            (nsum[dh] + EPSF);
  const float* hp = h + ((size_t)s * H + hd) * C;
  float* op = out + ((size_t)d * H + hd) * C;
  for (int c = 0; c < C; ++c) atomicAdd(&op[c], w * hp[c]);
}

// ---------------- bias + activation (0 = ELU, 1 = ReLU) ------------------------
__global__ void bias_act(float* __restrict__ x, const float* __restrict__ b, long n,
                         int F, int mode) {
  long i = blockIdx.x * (long)blockDim.x + threadIdx.x;
  if (i >= n) return;
  float v = x[i] + b[(int)(i % F)];
  x[i] = (mode == 0) ? (v > 0.f ? v : expf(v) - 1.f) : fmaxf(v, 0.f);
}

// ---------------- gate MLP second layer: gate[n] = g1[n,:] . Wg2 + bg2 ---------
__global__ void gate_dot(const float* __restrict__ g1, const float* __restrict__ Wg2,
                         const float* __restrict__ bg2, float* __restrict__ gate,
                         int N, int C) {
  int n = (int)(blockIdx.x * (long)blockDim.x + threadIdx.x);
  if (n >= N) return;
  float s = bg2[0];
  const float* gp = g1 + (size_t)n * C;
  for (int c = 0; c < C; ++c) s += gp[c] * Wg2[c];
  gate[n] = s;
}

// ---------------- graph-level segment softmax + weighted pooling ---------------
__global__ void graph_max(const float* __restrict__ gate, const long long* __restrict__ bm,
                          unsigned* __restrict__ gmaxk, int N) {
  int n = (int)(blockIdx.x * (long)blockDim.x + threadIdx.x);
  if (n >= N) return;
  atomicMax(&gmaxk[(int)bm[n]], fkey(gate[n]));
}
__global__ void graph_sum(const float* __restrict__ gate, const long long* __restrict__ bm,
                          const unsigned* __restrict__ gmaxk, float* __restrict__ gsum,
                          int N) {
  int n = (int)(blockIdx.x * (long)blockDim.x + threadIdx.x);
  if (n >= N) return;
  int g = (int)bm[n];
  atomicAdd(&gsum[g], expf(gate[n] - funkey(gmaxk[g])));
}
__global__ void graph_agg(const float* __restrict__ gate, const long long* __restrict__ bm,
                          const unsigned* __restrict__ gmaxk, const float* __restrict__ gsum,
                          const float* __restrict__ h2, float* __restrict__ out,
                          int N, int C) {
  int n = (int)(blockIdx.x * (long)blockDim.x + threadIdx.x);
  if (n >= N) return;
  int g = (int)bm[n];
  float w = expf(gate[n] - funkey(gmaxk[g])) / (gsum[g] + EPSF);
  const float* hp = h2 + (size_t)n * C;
  float* op = out + (size_t)g * C;
  for (int c = 0; c < C; ++c) atomicAdd(&op[c], w * hp[c]);
}

// ================================ host side ====================================
extern "C" void kernel_launch(void* const* d_in, const int* in_sizes, int n_in,
                              void* d_out, int out_size, void* d_ws, size_t ws_size,
                              hipStream_t stream) {
  const int   EMB = 128, HID = 64, HEADS = 4, HC1 = HEADS * HID, G = 64;
  const int   N = in_sizes[0] / EMB;     // 50000 (multiple of 16)
  const long  E = in_sizes[1] / 2;       // 800000
  const long  EL = E + N;                // with self loops

  const float*     x    = (const float*)d_in[0];
  const long long* ei   = (const long long*)d_in[1];
  const long long* bm   = (const long long*)d_in[2];
  const float*     W1   = (const float*)d_in[3];
  const float*     as1  = (const float*)d_in[4];
  const float*     ad1  = (const float*)d_in[5];
  const float*     b1   = (const float*)d_in[6];
  const float*     W2   = (const float*)d_in[7];
  const float*     as2  = (const float*)d_in[8];
  const float*     ad2  = (const float*)d_in[9];
  const float*     b2   = (const float*)d_in[10];
  const float*     Wg1  = (const float*)d_in[11];
  const float*     bg1  = (const float*)d_in[12];
  const float*     Wg2  = (const float*)d_in[13];
  const float*     bg2  = (const float*)d_in[14];
  float*           out  = (float*)d_out;

  // carve workspace
  size_t off = 0;
  auto carve = [&](size_t bytes) -> void* {
    void* p = (char*)d_ws + off;
    off += (bytes + 255) & ~(size_t)255;
    return p;
  };
  float*    h1     = (float*)carve((size_t)N * HC1 * 4);   // GEMM1 out (pre-agg features)
  float*    out1   = (float*)carve((size_t)N * HC1 * 4);   // layer1 aggregate -> layer2 input
  float*    a_src1 = (float*)carve((size_t)N * HEADS * 4);
  float*    a_dst1 = (float*)carve((size_t)N * HEADS * 4);
  unsigned* nmaxk1 = (unsigned*)carve((size_t)N * HEADS * 4);
  float*    nsum1  = (float*)carve((size_t)N * HEADS * 4);
  float*    h2     = (float*)carve((size_t)N * HID * 4);   // GEMM2 out
  float*    out2   = (float*)carve((size_t)N * HID * 4);   // layer2 aggregate -> node emb
  float*    a_src2 = (float*)carve((size_t)N * 4);
  float*    a_dst2 = (float*)carve((size_t)N * 4);
  unsigned* nmaxk2 = (unsigned*)carve((size_t)N * 4);
  float*    nsum2  = (float*)carve((size_t)N * 4);
  float*    g1     = (float*)carve((size_t)N * HID * 4);   // gate MLP hidden
  float*    gate   = (float*)carve((size_t)N * 4);
  unsigned* gmaxk  = (unsigned*)carve((size_t)G * 4);
  float*    gsum   = (float*)carve((size_t)G * 4);
  __bf16*   Ap1    = (__bf16*)carve((size_t)N * EMB * 2);  // packed bf16 fragments
  __bf16*   Bp1    = (__bf16*)carve((size_t)EMB * HC1 * 2);
  __bf16*   Ap2    = (__bf16*)carve((size_t)N * HC1 * 2);
  __bf16*   Bp2    = (__bf16*)carve((size_t)HC1 * HID * 2);
  __bf16*   Ap3    = (__bf16*)carve((size_t)N * HID * 2);
  __bf16*   Bp3    = (__bf16*)carve((size_t)HID * HID * 2);
  (void)ws_size; (void)n_in; (void)out_size;

  const int TB = 256;
  auto blocks = [&](long n) { return (unsigned)((n + TB - 1) / TB); };
  auto gemm_waves = [&](int M, int Ncols) { return (long)(M >> 4) * (Ncols >> 4); };

  // ---- init accumulators (every call; harness does not re-poison) ----
  fill_u32<<<blocks((long)G * HID), TB, 0, stream>>>((unsigned*)out, 0u, (long)G * HID);
  fill_u32<<<blocks((long)N * HC1), TB, 0, stream>>>((unsigned*)out1, 0u, (long)N * HC1);
  fill_u32<<<blocks((long)N * HID), TB, 0, stream>>>((unsigned*)out2, 0u, (long)N * HID);
  fill_u32<<<blocks((long)N * HEADS), TB, 0, stream>>>(nmaxk1, 0u, (long)N * HEADS);
  fill_u32<<<blocks((long)N * HEADS), TB, 0, stream>>>((unsigned*)nsum1, 0u, (long)N * HEADS);
  fill_u32<<<blocks(N), TB, 0, stream>>>(nmaxk2, 0u, N);
  fill_u32<<<blocks(N), TB, 0, stream>>>((unsigned*)nsum2, 0u, N);
  fill_u32<<<blocks(G), TB, 0, stream>>>(gmaxk, 0u, G);
  fill_u32<<<blocks(G), TB, 0, stream>>>((unsigned*)gsum, 0u, G);

  // ---- layer 1: h1 = x @ W1  (M=N, N=256, K=128) ----
  pack_A<<<blocks((long)N * EMB / 16), TB, 0, stream>>>(x, Ap1, N, EMB);
  pack_B<<<blocks((long)HC1 * EMB / 16), TB, 0, stream>>>(W1, Bp1, EMB, HC1);
  gemm_wmma_packed<128><<<blocks(gemm_waves(N, HC1) * 32), TB, 0, stream>>>(Ap1, Bp1, h1,
                                                                            N, HC1);
  attn_dot<<<blocks((long)N * HEADS), TB, 0, stream>>>(h1, as1, ad1, a_src1, a_dst1,
                                                       N, HEADS, HID);
  edge_attn_max<<<blocks(EL * HEADS), TB, 0, stream>>>(ei, E, N, HEADS, a_src1, a_dst1,
                                                       nmaxk1);
  edge_attn_sum<<<blocks(EL * HEADS), TB, 0, stream>>>(ei, E, N, HEADS, a_src1, a_dst1,
                                                       nmaxk1, nsum1);
  edge_aggregate<<<blocks(EL * HEADS), TB, 0, stream>>>(ei, E, N, HEADS, HID, a_src1,
                                                        a_dst1, nmaxk1, nsum1, h1, out1);
  bias_act<<<blocks((long)N * HC1), TB, 0, stream>>>(out1, b1, (long)N * HC1, HC1, 0);

  // ---- layer 2: h2 = out1 @ W2  (M=N, N=64, K=256), 1 head ----
  pack_A<<<blocks((long)N * HC1 / 16), TB, 0, stream>>>(out1, Ap2, N, HC1);
  pack_B<<<blocks((long)HID * HC1 / 16), TB, 0, stream>>>(W2, Bp2, HC1, HID);
  gemm_wmma_packed<256><<<blocks(gemm_waves(N, HID) * 32), TB, 0, stream>>>(Ap2, Bp2, h2,
                                                                            N, HID);
  attn_dot<<<blocks((long)N), TB, 0, stream>>>(h2, as2, ad2, a_src2, a_dst2, N, 1, HID);
  edge_attn_max<<<blocks(EL), TB, 0, stream>>>(ei, E, N, 1, a_src2, a_dst2, nmaxk2);
  edge_attn_sum<<<blocks(EL), TB, 0, stream>>>(ei, E, N, 1, a_src2, a_dst2, nmaxk2, nsum2);
  edge_aggregate<<<blocks(EL), TB, 0, stream>>>(ei, E, N, 1, HID, a_src2, a_dst2,
                                                nmaxk2, nsum2, h2, out2);
  bias_act<<<blocks((long)N * HID), TB, 0, stream>>>(out2, b2, (long)N * HID, HID, 0);

  // ---- attentional aggregation: g1 = relu(out2 @ Wg1 + bg1) ----
  pack_A<<<blocks((long)N * HID / 16), TB, 0, stream>>>(out2, Ap3, N, HID);
  pack_B<<<blocks((long)HID * HID / 16), TB, 0, stream>>>(Wg1, Bp3, HID, HID);
  gemm_wmma_packed<64><<<blocks(gemm_waves(N, HID) * 32), TB, 0, stream>>>(Ap3, Bp3, g1,
                                                                           N, HID);
  bias_act<<<blocks((long)N * HID), TB, 0, stream>>>(g1, bg1, (long)N * HID, HID, 1);
  gate_dot<<<blocks(N), TB, 0, stream>>>(g1, Wg2, bg2, gate, N, HID);
  graph_max<<<blocks(N), TB, 0, stream>>>(gate, bm, gmaxk, N);
  graph_sum<<<blocks(N), TB, 0, stream>>>(gate, bm, gmaxk, gsum, N);
  graph_agg<<<blocks(N), TB, 0, stream>>>(gate, bm, gmaxk, gsum, out2, out, N, HID);
}
